// model_2_76166950028524
// MI455X (gfx1250) — matrix operand order repaired and checked by reading, not compiled, not run on hardware
//
#include <hip/hip_runtime.h>
#include <hip/hip_bf16.h>
#include <math.h>

// Model dims
#define LQ 1024   // sequence length
#define HQ 256    // LSTM hidden per direction
#define EQ 384    // embedding width (300 + 84)
#define FQ 128    // E/3 after maxpool, also MLP_H

// ---------------------------------------------------------------------------
// CDNA5 WMMA types (wave32, 16x16x32 bf16 -> f32)
// ---------------------------------------------------------------------------
typedef __attribute__((ext_vector_type(16))) __bf16 v16bf;
typedef __attribute__((ext_vector_type(8)))  float  v8f;

union FragBF { v16bf v; unsigned int w[8]; };

__device__ __forceinline__ unsigned short f2bf(float f) {
  unsigned int u = __builtin_bit_cast(unsigned int, f);
  u += 0x7fffu + ((u >> 16) & 1u);           // round-to-nearest-even
  return (unsigned short)(u >> 16);
}
__device__ __forceinline__ float bflo2f(unsigned int p) {
  return __builtin_bit_cast(float, p << 16);          // low bf16 of dword
}
__device__ __forceinline__ float bfhi2f(unsigned int p) {
  return __builtin_bit_cast(float, p & 0xffff0000u);  // high bf16 of dword
}
__device__ __forceinline__ float sigmoidf(float x) {
  return 1.0f / (1.0f + __expf(-x));
}

// CDNA5 TRANS32 tanh if the toolchain exposes it, else libm.
#if __has_builtin(__builtin_amdgcn_tanhf)
#define TANHF(x) __builtin_amdgcn_tanhf(x)
#else
#define TANHF(x) tanhf(x)
#endif

// ---------------------------------------------------------------------------
// f32 -> bf16 conversion with optional row stride / column offset
// ---------------------------------------------------------------------------
__global__ __launch_bounds__(256) void cvt_bf16(
    const float* __restrict__ src, unsigned short* __restrict__ dst,
    long long n, int cols, int src_stride, int src_off)
{
  long long i = (long long)blockIdx.x * 256 + threadIdx.x;
  if (i >= n) return;
  long long r = i / cols;
  int c = (int)(i - r * cols);
  dst[i] = f2bf(src[r * (long long)src_stride + src_off + c]);
}

// ---------------------------------------------------------------------------
// Embedding gather + conv1d(5,pad2)+BN+ReLU x2 + maxpool(3) -> x bf16 (L x 128)
// ---------------------------------------------------------------------------
__global__ __launch_bounds__(128) void embed_conv_pool(
    const int* __restrict__ widx, const int* __restrict__ pidx,
    const float* __restrict__ wemb, const float* __restrict__ pemb,
    const float* __restrict__ c1w, const float* __restrict__ c1b,
    const float* __restrict__ c2w, const float* __restrict__ c2b,
    const float* __restrict__ g1w, const float* __restrict__ g1b,
    const float* __restrict__ g2w, const float* __restrict__ g2b,
    unsigned short* __restrict__ xbf)
{
  __shared__ float s0[EQ], s1[EQ];
  const int l = blockIdx.x, tid = threadIdx.x;
  const int wi = widx[l], pi = pidx[l];
  for (int e = tid; e < EQ; e += 128)
    s0[e] = (e < 300) ? wemb[(size_t)wi * 300 + e] : pemb[(size_t)pi * 84 + (e - 300)];
  __syncthreads();

  const float inv = 1.0f / sqrtf(1.0f + 1e-5f);
  {
    const float sc = g1w[0] * inv, sh = g1b[0], cb = c1b[0];
    float k0 = c1w[0], k1 = c1w[1], k2 = c1w[2], k3 = c1w[3], k4 = c1w[4];
    for (int e = tid; e < EQ; e += 128) {
      float acc = cb;
      if (e - 2 >= 0) acc += s0[e - 2] * k0;
      if (e - 1 >= 0) acc += s0[e - 1] * k1;
      acc += s0[e] * k2;
      if (e + 1 < EQ) acc += s0[e + 1] * k3;
      if (e + 2 < EQ) acc += s0[e + 2] * k4;
      s1[e] = fmaxf(acc * sc + sh, 0.0f);
    }
  }
  __syncthreads();
  {
    const float sc = g2w[0] * inv, sh = g2b[0], cb = c2b[0];
    float k0 = c2w[0], k1 = c2w[1], k2 = c2w[2], k3 = c2w[3], k4 = c2w[4];
    for (int e = tid; e < EQ; e += 128) {
      float acc = cb;
      if (e - 2 >= 0) acc += s1[e - 2] * k0;
      if (e - 1 >= 0) acc += s1[e - 1] * k1;
      acc += s1[e] * k2;
      if (e + 1 < EQ) acc += s1[e + 1] * k3;
      if (e + 2 < EQ) acc += s1[e + 2] * k4;
      s0[e] = fmaxf(acc * sc + sh, 0.0f);
    }
  }
  __syncthreads();
  if (tid < FQ) {
    float m = fmaxf(s0[3 * tid], fmaxf(s0[3 * tid + 1], s0[3 * tid + 2]));
    xbf[(size_t)l * FQ + tid] = f2bf(m);
  }
}

// ---------------------------------------------------------------------------
// WMMA GEMM: out[M x N] = X[M x K] * W[N x K]^T (+ bias[N]), bf16 in, f32 out.
// Block = 128 threads (4 waves); wave tile 16x64; block tile 64x64.
// K is a template parameter (128 or 512) so the pipeline loop has a
// compile-time trip count. The shared 64x32 B-tile is staged in LDS via
// GLOBAL_LOAD_ASYNC_TO_LDS_B128 (2 unconditional issues per wave per tile),
// 4-deep buffer rotation, retired with s_wait_asynccnt 2 so the copy of
// tile i+1 overlaps WMMA on tile i. Tile i+1 is prefetched unconditionally;
// the one garbage tile read past K lands inside later workspace regions and
// is never consumed (S_ENDPGM's implicit wait-idle retires it).
// LDS rows padded to 80B to spread b128 reads across banks.
// ---------------------------------------------------------------------------
#define BROW_PAD 40   // u16 per LDS B-tile row (80 bytes, 16B-aligned)

template <int K>
__global__ __launch_bounds__(128) void wmma_gemm_bf16nt(
    const unsigned short* __restrict__ X,
    const unsigned short* __restrict__ W,
    const float* __restrict__ bias,
    float* __restrict__ out,
    int N)
{
  __shared__ unsigned short sB[4][64 * BROW_PAD];   // 4 x 5KB
  const int tid  = threadIdx.x;
  const int lane = tid & 31;
  const int wave = tid >> 5;
  const int hl   = lane >> 4;   // lane half (0: lanes 0-15, 1: lanes 16-31)
  const int lr   = lane & 15;
  const int m0 = blockIdx.y * 64 + wave * 16;
  const int n0 = blockIdx.x * 64;

  // Per-thread chunk coordinates for the cooperative B-tile copy:
  // chunk c (16B) -> row = c>>2, 16B-column = c&3. Thread owns c=tid, tid+128.
  const int r0 = tid >> 2, c4 = tid & 3;

  // Issue the two async lane-copies for tile (k0) into buffer `buf`.
  auto issueB = [&](int buf, int k0) {
    unsigned long long g0 = (unsigned long long)
        (W + (size_t)(n0 + r0) * K + k0 + c4 * 8);
    unsigned int l0 = (unsigned int)(unsigned long long)
        (&sB[buf][r0 * BROW_PAD + c4 * 8]);   // low 32 bits = LDS offset
    asm volatile("global_load_async_to_lds_b128 %0, %1, off"
                 :: "v"(l0), "v"(g0) : "memory");
    unsigned long long g1 = (unsigned long long)
        (W + (size_t)(n0 + 32 + r0) * K + k0 + c4 * 8);
    unsigned int l1 = (unsigned int)(unsigned long long)
        (&sB[buf][(32 + r0) * BROW_PAD + c4 * 8]);
    asm volatile("global_load_async_to_lds_b128 %0, %1, off"
                 :: "v"(l1), "v"(g1) : "memory");
  };

  v8f acc[4] = {};
  const unsigned short* xrow = X + (size_t)(m0 + lr) * K;
  constexpr int nsteps = K / 32;

  issueB(0, 0);
#pragma unroll 4
  for (int i = 0; i < nsteps; ++i) {
    issueB((i + 1) & 3, (i + 1) * 32);                 // prefetch (may overrun)
    asm volatile("s_wait_asynccnt 0x2" ::: "memory");  // tile i complete
    __syncthreads();   // all waves' async copies of tile i visible

    const int k0 = i * 32;
    // A fragment: row m=lr; elems 0..7 -> k0+hl*8+i ; 8..15 -> k0+16+hl*8+i
    FragBF a;
    const unsigned int* a0 = (const unsigned int*)(xrow + k0 + hl * 8);
    const unsigned int* a1 = (const unsigned int*)(xrow + k0 + 16 + hl * 8);
#pragma unroll
    for (int r = 0; r < 4; ++r) { a.w[r] = a0[r]; a.w[4 + r] = a1[r]; }

    const unsigned short* bt = sB[i & 3];
#pragma unroll
    for (int s = 0; s < 4; ++s) {
      // B fragment from LDS: lane(hl,lr) reads row n=s*16+lr, k = hl*8..+7, 16+hl*8..+7
      FragBF b;
      const unsigned int* bl =
          (const unsigned int*)(bt + (s * 16 + lr) * BROW_PAD + hl * 8);
#pragma unroll
      for (int r = 0; r < 4; ++r) { b.w[r] = bl[r]; b.w[4 + r] = bl[8 + r]; }
      acc[s] = __builtin_amdgcn_wmma_f32_16x16x32_bf16(
          false, a.v, false, b.v, (short)0, acc[s], false, false);
    }
  }

#pragma unroll
  for (int s = 0; s < 4; ++s) {
    const int n = n0 + s * 16 + lr;
    const float bv = bias ? bias[n] : 0.0f;
#pragma unroll
    for (int r = 0; r < 8; ++r)                   // D: (m = hl*8 + r, n = lr)
      out[(size_t)(m0 + hl * 8 + r) * N + n] = acc[s][r] + bv;
  }
}

// ---------------------------------------------------------------------------
// LSTM recurrence (serial scan). One block per direction (2 blocks total).
// xw holds x@Wih^T + b for all timesteps (WMMA-precomputed), so each step is
// only the 1024x256 recurrent matvec. The i/f-gate half of Whh (512 rows,
// 256KB bf16) is pinned in LDS (gfx1250: 320KB/workgroup); the g/o half
// streams from L2. Gate order per jnp.split: [i | f | g | o].
// ---------------------------------------------------------------------------
__global__ __launch_bounds__(256) void lstm_recur(
    const float* __restrict__ xw,           // [2][L][1024], dir-major
    const unsigned short* __restrict__ whh, // [2][1024][256] bf16
    unsigned short* __restrict__ hout,      // [L][512] bf16 (dir d -> cols d*256..)
    int T)
{
  __shared__ unsigned short wlds[512 * HQ];      // 256KB: Whh rows 0..511
  __shared__ float h[HQ], c[HQ], gates[4 * HQ];  // 6KB
  const int dir = blockIdx.x;
  const int tid = threadIdx.x;
  const float* xwd = xw + (size_t)dir * T * (4 * HQ);
  const unsigned short* whd = whh + (size_t)dir * (4 * HQ) * HQ;

  // Stage rows 0..511 of Whh into LDS (16384 x 16B chunks).
  {
    const uint4* src = (const uint4*)whd;
    uint4* dst = (uint4*)wlds;
    for (int i = tid; i < (512 * HQ * 2) / 16; i += 256) dst[i] = src[i];
  }
  h[tid] = 0.0f;
  c[tid] = 0.0f;
  __syncthreads();

  int t = dir ? (T - 1) : 0;
  const int step = dir ? -1 : 1;
  for (int it = 0; it < T; ++it, t += step) {
    const float* xrow = xwd + (size_t)t * (4 * HQ);

    // i/f gates: weights from LDS
#pragma unroll
    for (int q = 0; q < 2; ++q) {
      const int j = tid + q * HQ;
      const unsigned int* wr = (const unsigned int*)(wlds + (size_t)j * HQ);
      float acc = xrow[j];
#pragma unroll 8
      for (int kk = 0; kk < HQ / 2; ++kk) {
        const unsigned int p = wr[kk];
        acc = fmaf(bflo2f(p), h[2 * kk], acc);
        acc = fmaf(bfhi2f(p), h[2 * kk + 1], acc);
      }
      gates[j] = acc;
    }
    // g/o gates: weights streamed from L2
#pragma unroll
    for (int q = 2; q < 4; ++q) {
      const int j = tid + q * HQ;
      const unsigned int* wr = (const unsigned int*)(whd + (size_t)j * HQ);
      float acc = xrow[j];
#pragma unroll 8
      for (int kk = 0; kk < HQ / 2; ++kk) {
        const unsigned int p = wr[kk];
        acc = fmaf(bflo2f(p), h[2 * kk], acc);
        acc = fmaf(bfhi2f(p), h[2 * kk + 1], acc);
      }
      gates[j] = acc;
    }
    __syncthreads();

    const float ig = sigmoidf(gates[tid]);
    const float fg = sigmoidf(gates[tid + HQ]);
    const float gg = TANHF(gates[tid + 2 * HQ]);
    const float og = sigmoidf(gates[tid + 3 * HQ]);
    const float cn = fg * c[tid] + ig * gg;
    const float hn = og * TANHF(cn);
    c[tid] = cn;
    h[tid] = hn;
    hout[(size_t)t * (2 * HQ) + dir * HQ + tid] = f2bf(hn);
    __syncthreads();
  }
}

// ---------------------------------------------------------------------------
// Pair scorer: out[i][j] = b2 + sum_f w2[f]*tanh(A[i][f]+B[j][f]+b1[f]);
// diagonal forced to 0. 32x32 score tile per block, A/B rows staged in LDS.
// ---------------------------------------------------------------------------
__global__ __launch_bounds__(256) void pair_score(
    const float* __restrict__ A, const float* __restrict__ Bm,
    const float* __restrict__ b1, const float* __restrict__ w2,
    const float* __restrict__ b2, float* __restrict__ out)
{
  __shared__ float sA[32][FQ], sB2[32][FQ], sb[FQ], sw[FQ];
  const int i0 = blockIdx.y * 32, j0 = blockIdx.x * 32;
  for (int tI = threadIdx.x; tI < 32 * FQ; tI += 256) {
    const int r = tI >> 7, cI = tI & 127;
    sA[r][cI]  = A[(size_t)(i0 + r) * FQ + cI];
    sB2[r][cI] = Bm[(size_t)(j0 + r) * FQ + cI];
  }
  if (threadIdx.x < FQ) {
    sb[threadIdx.x] = b1[threadIdx.x];
    sw[threadIdx.x] = w2[threadIdx.x];
  }
  __syncthreads();

  const float bb = b2[0];
  for (int p = threadIdx.x; p < 1024; p += 256) {
    const int ii = p >> 5, jj = p & 31;
    float acc = bb;
#pragma unroll 4
    for (int f = 0; f < FQ; ++f)
      acc += sw[f] * TANHF(sA[ii][f] + sB2[jj][f] + sb[f]);
    const int gi = i0 + ii, gj = j0 + jj;
    out[(size_t)gi * LQ + gj] = (gi == gj) ? 0.0f : acc;
  }
}

// ---------------------------------------------------------------------------
// Host orchestration
// ---------------------------------------------------------------------------
extern "C" void kernel_launch(void* const* d_in, const int* in_sizes, int n_in,
                              void* d_out, int out_size, void* d_ws, size_t ws_size,
                              hipStream_t stream) {
  (void)in_sizes; (void)n_in; (void)out_size; (void)ws_size;

  const int*   word_idx = (const int*)d_in[0];
  const int*   pos_idx  = (const int*)d_in[1];
  const float* wemb = (const float*)d_in[3];
  const float* pemb = (const float*)d_in[4];
  const float* c1w = (const float*)d_in[5];  const float* c1b = (const float*)d_in[6];
  const float* c2w = (const float*)d_in[7];  const float* c2b = (const float*)d_in[8];
  const float* g1w = (const float*)d_in[9];  const float* g1b = (const float*)d_in[10];
  const float* g2w = (const float*)d_in[11]; const float* g2b = (const float*)d_in[12];
  const float* WihF[4] = {(const float*)d_in[13], (const float*)d_in[16],
                          (const float*)d_in[19], (const float*)d_in[22]};
  const float* WhhF[4] = {(const float*)d_in[14], (const float*)d_in[17],
                          (const float*)d_in[20], (const float*)d_in[23]};
  const float* Bg[4]   = {(const float*)d_in[15], (const float*)d_in[18],
                          (const float*)d_in[21], (const float*)d_in[24]};
  const float* W1 = (const float*)d_in[25];
  const float* b1 = (const float*)d_in[26];
  const float* w2 = (const float*)d_in[27];
  const float* b2 = (const float*)d_in[28];
  const int Kin[4] = {128, 512, 512, 512};

  // Workspace layout (256B aligned; total ~22.5 MB, L2-resident)
  size_t off = 0;
  auto alloc = [&](size_t bytes) -> void* {
    void* p = (char*)d_ws + off;
    off += (bytes + 255) & ~(size_t)255;
    return p;
  };
  unsigned short* xbf = (unsigned short*)alloc((size_t)LQ * FQ * 2);
  unsigned short* wih[4];
  unsigned short* whh[4];
  for (int l = 0; l < 4; ++l) wih[l] = (unsigned short*)alloc((size_t)2 * 1024 * Kin[l] * 2);
  for (int l = 0; l < 4; ++l) whh[l] = (unsigned short*)alloc((size_t)2 * 1024 * HQ * 2);
  unsigned short* whA = (unsigned short*)alloc((size_t)FQ * 512 * 2);
  unsigned short* whB = (unsigned short*)alloc((size_t)FQ * 512 * 2);
  float* xw = (float*)alloc((size_t)2 * LQ * 1024 * 4);
  unsigned short* hb[2];
  hb[0] = (unsigned short*)alloc((size_t)LQ * 512 * 2);
  hb[1] = (unsigned short*)alloc((size_t)LQ * 512 * 2);
  float* Ap = (float*)alloc((size_t)LQ * FQ * 4);
  float* Bp = (float*)alloc((size_t)LQ * FQ * 4);

  // 1) Front end: gather + conv/BN/ReLU x2 + maxpool -> x bf16
  embed_conv_pool<<<LQ, 128, 0, stream>>>(word_idx, pos_idx, wemb, pemb,
                                          c1w, c1b, c2w, c2b, g1w, g1b, g2w, g2b, xbf);

  // 2) bf16 weight copies
  auto cvt = [&](const float* src, unsigned short* dst, long long n,
                 int cols, int stride, int coff) {
    cvt_bf16<<<(int)((n + 255) / 256), 256, 0, stream>>>(src, dst, n, cols, stride, coff);
  };
  for (int l = 0; l < 4; ++l) {
    cvt(WihF[l], wih[l], 2LL * 1024 * Kin[l], Kin[l], Kin[l], 0);
    cvt(WhhF[l], whh[l], 2LL * 1024 * HQ, HQ, HQ, 0);
  }
  cvt(W1, whA, (long long)FQ * 512, 512, 1024, 0);    // Wh = mlp_W1[:, :512]
  cvt(W1, whB, (long long)FQ * 512, 512, 1024, 512);  // Wm = mlp_W1[:, 512:]

  // 3) 4 BiLSTM layers: WMMA xw precompute + serial scan per direction
  const unsigned short* inb = xbf;
  for (int l = 0; l < 4; ++l) {
    for (int d = 0; d < 2; ++d) {
      if (Kin[l] == 128) {
        wmma_gemm_bf16nt<128><<<dim3(1024 / 64, LQ / 64), 128, 0, stream>>>(
            inb, wih[l] + (size_t)d * 1024 * 128, Bg[l] + (size_t)d * 1024,
            xw + (size_t)d * LQ * 1024, 1024);
      } else {
        wmma_gemm_bf16nt<512><<<dim3(1024 / 64, LQ / 64), 128, 0, stream>>>(
            inb, wih[l] + (size_t)d * 1024 * 512, Bg[l] + (size_t)d * 1024,
            xw + (size_t)d * LQ * 1024, 1024);
      }
    }
    unsigned short* hout = hb[l & 1];
    lstm_recur<<<2, 256, 0, stream>>>(xw, whh[l], hout, LQ);
    inb = hout;
  }

  // 4) Biaffine projections (WMMA) + pair scorer
  wmma_gemm_bf16nt<512><<<dim3(FQ / 64, LQ / 64), 128, 0, stream>>>(inb, whA, nullptr, Ap, FQ);
  wmma_gemm_bf16nt<512><<<dim3(FQ / 64, LQ / 64), 128, 0, stream>>>(inb, whB, nullptr, Bp, FQ);
  pair_score<<<dim3(LQ / 32, LQ / 32), 256, 0, stream>>>(Ap, Bp, b1, w2, b2, (float*)d_out);
}
